// GroundedRefinementBlock_65300682769110
// MI455X (gfx1250) — matrix operand order repaired and checked by reading, not compiled, not run on hardware
//
#include <hip/hip_runtime.h>
#include <hip/hip_bf16.h>

// ---------------------------------------------------------------------------
// GroundedRefinementBlock for MI455X (gfx1250), wave32 + WMMA bf16 + TDM.
// All 1x1 convs and the feature-space Gram matrix run through one generic
// WMMA GEMM kernel (v_wmma_f32_16x16x32_bf16). Operand panels are staged to
// LDS (tensor_load_to_lds when available, coalesced b128 loads otherwise);
// all K/M tails are zero-padded so the inner loop is branch-free.
// ---------------------------------------------------------------------------

#define BB    4
#define NN    2048
#define FEATC 256
#define DIMC  128
#define NKNN  16
#define KI    8
#define UPF   2
#define POSH  64
#define ATTNH 512
#define QDIM  515    // 3 + 2*FEAT
#define QDIMP 544    // QDIM padded to multiple of 32
#define POSKP 32     // 3 padded to 32
#define MAXKP 544    // largest padded K

typedef __attribute__((ext_vector_type(16))) __bf16 v16bf;
typedef __attribute__((ext_vector_type(8)))  float  v8f;

#if defined(__has_builtin)
#if __has_builtin(__builtin_amdgcn_tensor_load_to_lds) && \
    __has_builtin(__builtin_amdgcn_s_wait_tensorcnt)
#define USE_TDM 1
#endif
#endif
#ifndef USE_TDM
#define USE_TDM 0
#endif

// Stage a [Kp x 16] bf16 panel (row stride krs elements, 16 contiguous
// columns) from global memory into LDS at s (row-major, row stride 16).
__device__ __forceinline__ void stage_panel(const __bf16* __restrict__ g, long krs,
                                            __bf16* s, int Kp) {
#if USE_TDM
  typedef __attribute__((ext_vector_type(4))) unsigned int u32x4;
  typedef __attribute__((ext_vector_type(8))) int i32x8;
  typedef __attribute__((ext_vector_type(4))) int i32x4;
  unsigned long long ga = (unsigned long long)g;
  unsigned int lds = (unsigned int)(unsigned long long)s;  // low 32 = LDS offset
  u32x4 g0; i32x8 g1; i32x4 g2; i32x4 g3; i32x8 g4;
  g0[0] = 1u;                                        // count=1 (valid user D#)
  g0[1] = lds;                                       // lds_addr (bytes)
  g0[2] = (unsigned int)(ga & 0xFFFFFFFFull);        // global_addr[31:0]
  g0[3] = (unsigned int)((ga >> 32) & 0x01FFFFFFull) // global_addr[56:32]
          | (2u << 30);                              // type = 2 (image)
  g1[0] = 0x00010000;                                // data_size=1 (2B), mask=0
  g1[1] = (int)(16u << 16);                          // tensor_dim0 = 16
  g1[2] = (int)(((unsigned)Kp & 0xFFFFu) << 16);     // tensor_dim1 = Kp (lo16)
  g1[3] = (int)(16u << 16);                          // dim1 hi=0 | tile_dim0=16
  g1[4] = (int)((unsigned)Kp & 0xFFFFu);             // tile_dim1 = Kp, tile_dim2=0
  g1[5] = (int)(unsigned)krs;                        // tensor_dim0_stride lo32
  g1[6] = 0;                                         // stride hi | dim1_stride lo
  g1[7] = 0;
  g2[0] = g2[1] = g2[2] = g2[3] = 0;                 // 2D tensor: groups 2/3 unused
  g3[0] = g3[1] = g3[2] = g3[3] = 0;
  g4[0] = g4[1] = g4[2] = g4[3] = 0;
  g4[4] = g4[5] = g4[6] = g4[7] = 0;
  __builtin_amdgcn_tensor_load_to_lds(g0, g1, g2, g3, g4, 0);
#else
  int lane = threadIdx.x;
  for (int k = lane; k < Kp; k += 32) {
    const uint4* src = reinterpret_cast<const uint4*>(g + (long)k * krs);
    uint4 lo = src[0];
    uint4 hi = src[1];
    uint4* dst = reinterpret_cast<uint4*>(s + (long)k * 16);
    dst[0] = lo;
    dst[1] = hi;
  }
#endif
}

// ------------------------------ elementwise --------------------------------

__global__ void k_cvt(const float* __restrict__ src, __bf16* __restrict__ dst,
                      long n, int relu) {
  long t = (long)blockIdx.x * blockDim.x + threadIdx.x;
  if (t >= n) return;
  float v = src[t];
  if (relu) v = fmaxf(v, 0.f);
  dst[t] = (__bf16)v;
}

// weight (O,C) f32 -> zero-padded (Op,Cp) bf16
__global__ void k_cvt_pad(const float* __restrict__ src, __bf16* __restrict__ dst,
                          int O, int C, int Op, int Cp) {
  long t = (long)blockIdx.x * blockDim.x + threadIdx.x;
  if (t >= (long)Op * Cp) return;
  int c = (int)(t % Cp), o = (int)(t / Cp);
  float v = (o < O && c < C) ? src[(long)o * C + c] : 0.f;
  dst[t] = (__bf16)v;
}

__global__ void k_add(float* __restrict__ dst, const float* __restrict__ src, long n) {
  long t = (long)blockIdx.x * blockDim.x + threadIdx.x;
  if (t >= n) return;
  dst[t] += src[t];
}

// ------------------------------ WMMA GEMM ----------------------------------
// C[b, m, n] = sum_k A[b, m, k] * B[b, k, n]   (columns of B/C contiguous)
// ACONTIG: A is row-major with contiguous k (weights).   a_rs = padded row len
// !ACONTIG: A is k-major (strided rows, contiguous m).   a_cs = k stride
// Kp must be a multiple of 32 and all padded memory must be readable (zeros).
template <bool ACONTIG>
__global__ void k_wmma_gemm(
    const __bf16* __restrict__ A, long a_rs, long a_cs, long a_bs,
    const __bf16* __restrict__ Bm, long b_rs, long b_bs,
    float* __restrict__ C, long c_rs, long c_bs,
    __bf16* __restrict__ Cbf, long cb_rs, long cb_bs,
    const float* __restrict__ bias,
    const float* __restrict__ bn_g, const float* __restrict__ bn_bt,
    const float* __restrict__ bn_mean, const float* __restrict__ bn_rstd,
    float* __restrict__ stat_sum, float* __restrict__ stat_ssq,
    int M, int Kp, int flags)  // flags bit0: relu, bit1: accumulate f32
{
  __shared__ __bf16 sB[MAXKP * 16];
  __shared__ __bf16 sA[(ACONTIG ? 1 : MAXKP) * 16];

  const int lane = threadIdx.x;          // blockDim.x == 32 (one wave)
  const int half = lane >> 4;
  const int l16  = lane & 15;
  const long n0  = (long)blockIdx.x * 16;
  const int  m0  = blockIdx.y * 16;
  const int  b   = blockIdx.z;
  const __bf16* Ab = A + (long)b * a_bs;
  const __bf16* Bb = Bm + (long)b * b_bs;

  // stage panels into LDS
  stage_panel(Bb + n0, b_rs, sB, Kp);
  if constexpr (!ACONTIG) stage_panel(Ab + m0, a_cs, sA, Kp);
#if USE_TDM
  __builtin_amdgcn_s_wait_tensorcnt(0);
#endif
  asm volatile("" ::: "memory");

  v8f acc = {};
  const int am = m0 + l16;               // A-fragment row for this lane
  const long bn = n0 + l16;              // B-fragment column for this lane

  for (int kk = 0; kk < Kp; kk += 32) {
    v16bf af, bfr;
    if constexpr (ACONTIG) {
      // two 128-bit loads: k = kk+half*8 .. +7 and +16 .. +23
      const uint4* ap =
          reinterpret_cast<const uint4*>(Ab + (long)am * a_rs + kk + half * 8);
      union { uint4 u[2]; v16bf v; } au;
      au.u[0] = ap[0];
      au.u[1] = ap[2];
      af = au.v;
    } else {
#pragma unroll
      for (int e = 0; e < 8; ++e) {
        int k0 = kk + half * 8 + e;
        af[e]     = sA[(long)k0 * 16 + l16];
        af[e + 8] = sA[(long)(k0 + 16) * 16 + l16];
      }
    }
#pragma unroll
    for (int e = 0; e < 16; ++e) {
      int k = kk + half * 8 + (e & 7) + ((e >> 3) << 4);  // lanes 0-15: K 0..7,16..23; lanes 16-31: 8..15,24..31
      bfr[e] = sB[(long)k * 16 + l16];
    }
    acc = __builtin_amdgcn_wmma_f32_16x16x32_bf16(
        false, af, false, bfr, (short)0, acc, false, false);
  }

#pragma unroll
  for (int r = 0; r < 8; ++r) {
    int m = m0 + r + half * 8;           // C layout: n = lane%16, row = r + 8*(lane/16)
    if (m >= M) continue;
    float v = acc[r];
    if (bias)  v += bias[m];
    if (bn_g)  v = bn_g[m] * (v - bn_mean[m]) * bn_rstd[m] + bn_bt[m];
    if (flags & 1) v = fmaxf(v, 0.f);
    if (stat_sum) { atomicAdd(&stat_sum[m], v); atomicAdd(&stat_ssq[m], v * v); }
    if (C) {
      long ci = (long)b * c_bs + (long)m * c_rs + bn;
      if (flags & 2) C[ci] += v; else C[ci] = v;
    }
    if (Cbf) Cbf[(long)b * cb_bs + (long)m * cb_rs + bn] = (__bf16)v;
  }
}

// ------------------------------ kNN (3-D) ----------------------------------
__global__ void k_knn3(const float* __restrict__ q, const float* __restrict__ r,
                       int* __restrict__ oidx, float* __restrict__ ow,
                       int K, int useW) {
  long t = (long)blockIdx.x * blockDim.x + threadIdx.x;
  if (t >= (long)BB * NN) return;
  int b = (int)(t / NN), n = (int)(t % NN);
  const float* qb = q + (long)b * 3 * NN;
  const float* rb = r + (long)b * 3 * NN;
  float qx = qb[n], qy = qb[NN + n], qz = qb[2 * NN + n];
  float bd[NKNN]; int bi[NKNN];
  for (int i = 0; i < K; ++i) { bd[i] = 3.4e38f; bi[i] = 0; }
  for (int m = 0; m < NN; ++m) {
    float dx = qx - rb[m], dy = qy - rb[NN + m], dz = qz - rb[2 * NN + m];
    float d = dx * dx + dy * dy + dz * dz;
    if (d < bd[K - 1]) {
      int p = K - 1;
      while (p > 0 && bd[p - 1] > d) { bd[p] = bd[p - 1]; bi[p] = bi[p - 1]; --p; }
      bd[p] = d; bi[p] = m;
    }
  }
  int* oi = oidx + t * K;
  for (int i = 0; i < K; ++i) oi[i] = bi[i];
  if (useW) {
    float* wv = ow + t * K;
    float s = 0.f;
    for (int i = 0; i < K; ++i) {
      float d = fmaxf(bd[i], 0.f);
      float w = 1.f / (d + 1e-8f);
      wv[i] = w; s += w;
    }
    for (int i = 0; i < K; ++i) wv[i] /= s;
  }
}

// |x|^2 per point: x (B,C,N) -> out (B,N)
__global__ void k_rownorm(const float* __restrict__ x, float* __restrict__ out, int C) {
  long t = (long)blockIdx.x * blockDim.x + threadIdx.x;
  if (t >= (long)BB * NN) return;
  int b = (int)(t / NN), n = (int)(t % NN);
  const float* xb = x + (long)b * C * NN + n;
  float s = 0.f;
  for (int c = 0; c < C; ++c) { float v = xb[(long)c * NN]; s += v * v; }
  out[t] = s;
}

// top-k (k=KI) smallest of d2 = qn[q] + rn[m] - 2*S[q,m], one batch per launch
__global__ void k_topk_S(const float* __restrict__ S, const float* __restrict__ qn,
                         const float* __restrict__ rn, int* __restrict__ oidx,
                         float* __restrict__ ow) {
  int n = blockIdx.x * blockDim.x + threadIdx.x;
  if (n >= NN) return;
  float qv = qn[n];
  const float* Sr = S + (long)n * NN;
  float bd[KI]; int bi[KI];
  for (int i = 0; i < KI; ++i) { bd[i] = 3.4e38f; bi[i] = 0; }
  for (int m = 0; m < NN; ++m) {
    float d = qv + rn[m] - 2.f * Sr[m];
    if (d < bd[KI - 1]) {
      int p = KI - 1;
      while (p > 0 && bd[p - 1] > d) { bd[p] = bd[p - 1]; bi[p] = bi[p - 1]; --p; }
      bd[p] = d; bi[p] = m;
    }
  }
  int* oi = oidx + (long)n * KI;
  float* wv = ow + (long)n * KI;
  float s = 0.f;
  for (int i = 0; i < KI; ++i) {
    oi[i] = bi[i];
    float d = fmaxf(bd[i], 0.f);
    float w = 1.f / (d + 1e-8f);
    wv[i] = w; s += w;
  }
  for (int i = 0; i < KI; ++i) wv[i] /= s;
}

// IDW gather: out[b,c,n] = sum_k feat[b,c,idx[b,n,k]] * w[b,n,k]
__global__ void k_interp(const float* __restrict__ feat, const int* __restrict__ idx,
                         const float* __restrict__ wgt, float* __restrict__ out, int C) {
  long t = (long)blockIdx.x * blockDim.x + threadIdx.x;
  if (t >= (long)BB * C * NN) return;
  int n = (int)(t % NN);
  long bc = t / NN;
  int c = (int)(bc % C), b = (int)(bc / C);
  const int*   id = idx + ((long)b * NN + n) * KI;
  const float* wv = wgt + ((long)b * NN + n) * KI;
  const float* fb = feat + ((long)b * C + c) * (long)NN;
  float s = 0.f;
  for (int k = 0; k < KI; ++k) s += fb[id[k]] * wv[k];
  out[t] = s;
}

// q_raw concat -> f32 out (515 rows) + zero-padded bf16 copy (QDIMP rows)
__global__ void k_concat_qraw(const float* __restrict__ xyz,
                              const float* __restrict__ pi,
                              const float* __restrict__ gi,
                              float* __restrict__ oq, __bf16* __restrict__ oqb) {
  long t = (long)blockIdx.x * blockDim.x + threadIdx.x;
  if (t >= (long)BB * QDIMP * NN) return;
  int n = (int)(t % NN);
  long bc = t / NN;
  int c = (int)(bc % QDIMP), b = (int)(bc / QDIMP);
  float v = 0.f;
  if (c < 3)              v = xyz[((long)b * 3 + c) * NN + n];
  else if (c < 3 + FEATC) v = pi[((long)b * FEATC + (c - 3)) * NN + n];
  else if (c < QDIM)      v = gi[((long)b * FEATC + (c - 3 - FEATC)) * NN + n];
  oqb[t] = (__bf16)v;
  if (c < QDIM) oq[((long)b * QDIM + c) * NN + n] = v;
}

__global__ void k_finalize(const float* __restrict__ sum, const float* __restrict__ ssq,
                           float* __restrict__ mean, float* __restrict__ rstd,
                           int C, float cnt) {
  int c = blockIdx.x * blockDim.x + threadIdx.x;
  if (c >= C) return;
  float m = sum[c] / cnt;
  float v = ssq[c] / cnt - m * m;
  mean[c] = m;
  rstd[c] = rsqrtf(v + 1e-5f);
}

// pos_rel (B,POSKP,N,K) zero-padded bf16; rows 0..2 = xyz[n]-xyz[idx]
__global__ void k_posrel(const float* __restrict__ xyz, const int* __restrict__ idx,
                         __bf16* __restrict__ out) {
  long t = (long)blockIdx.x * blockDim.x + threadIdx.x;
  if (t >= (long)BB * POSKP * NN * NKNN) return;
  int k = (int)(t % NKNN);
  long r = t / NKNN;
  int n = (int)(r % NN);
  long bc = r / NN;
  int c = (int)(bc % POSKP), b = (int)(bc / POSKP);
  float v = 0.f;
  if (c < 3) {
    int j = idx[((long)b * NN + n) * NKNN + k];
    v = xyz[((long)b * 3 + c) * NN + n] - xyz[((long)b * 3 + c) * NN + j];
  }
  out[t] = (__bf16)v;
}

// qkpe[b,c,n,k] = q[b,c,n] - key[b,c,idx[b,n,k]] + pe[b,c,n,k]  (bf16)
__global__ void k_qkpe(const float* __restrict__ qm, const float* __restrict__ km,
                       const int* __restrict__ idx, const float* __restrict__ pe,
                       __bf16* __restrict__ out) {
  long t = (long)blockIdx.x * blockDim.x + threadIdx.x;
  if (t >= (long)BB * DIMC * NN * NKNN) return;
  int k = (int)(t % NKNN);
  long r = t / NKNN;
  int n = (int)(r % NN);
  long bc = r / NN;
  int c = (int)(bc % DIMC), b = (int)(bc / DIMC);
  int j = idx[((long)b * NN + n) * NKNN + k];
  float v = qm[((long)b * DIMC + c) * NN + n] - km[((long)b * DIMC + c) * NN + j] + pe[t];
  out[t] = (__bf16)v;
}

// softmax over K and aggregate with (v_g + pe); one batch per launch
__global__ void k_smax_agg(const float* __restrict__ logits, const float* __restrict__ vm,
                           const float* __restrict__ pe, const int* __restrict__ idx,
                           float* __restrict__ agg) {
  long t = (long)blockIdx.x * blockDim.x + threadIdx.x;
  if (t >= (long)DIMC * NN) return;
  int n = (int)(t % NN), c = (int)(t / NN);
  const float* L = logits + ((long)c * NN + n) * NKNN;
  float mx = -3.4e38f;
  for (int k = 0; k < NKNN; ++k) mx = fmaxf(mx, L[k]);
  float e[NKNN]; float s = 0.f;
  for (int k = 0; k < NKNN; ++k) { e[k] = __expf(L[k] - mx); s += e[k]; }
  const float* peb = pe + ((long)c * NN + n) * NKNN;
  float r = 0.f;
  for (int k = 0; k < NKNN; ++k) {
    int j = idx[(long)n * NKNN + k];
    r += (e[k] / s) * (vm[(long)c * NN + j] + peb[k]);
  }
  agg[t] = r;
}

// f2 = concat([f1, H], axis=1) -> f32 out + bf16 copy
__global__ void k_concat_f2(const float* __restrict__ f1, const float* __restrict__ H,
                            float* __restrict__ of, __bf16* __restrict__ ofb) {
  long t = (long)blockIdx.x * blockDim.x + threadIdx.x;
  if (t >= (long)BB * 2 * FEATC * NN) return;
  int n = (int)(t % NN);
  long bc = t / NN;
  int c = (int)(bc % (2 * FEATC)), b = (int)(bc / (2 * FEATC));
  float v = (c < FEATC) ? f1[((long)b * FEATC + c) * NN + n]
                        : H[((long)b * FEATC + (c - FEATC)) * NN + n];
  of[t] = v;
  ofb[t] = (__bf16)v;
}

// xyz_up[b,c,n*UP+u] = xyz[b,c,n] + dp[b, c*UP+u, n]
__global__ void k_shuffle(const float* __restrict__ xyz, const float* __restrict__ dp,
                          float* __restrict__ out) {
  long t = (long)blockIdx.x * blockDim.x + threadIdx.x;
  if (t >= (long)BB * 3 * NN * UPF) return;
  int u = (int)(t % UPF);
  long r = t / UPF;
  int n = (int)(r % NN);
  long bc = r / NN;
  int c = (int)(bc % 3), b = (int)(bc / 3);
  out[t] = xyz[((long)b * 3 + c) * NN + n] + dp[((long)b * 6 + c * UPF + u) * NN + n];
}

// ------------------------------ host side ----------------------------------

// jax pytree flatten order (dict keys sorted at every level)
enum {
  IN_GEN_FEAT = 0, IN_GEN_XYZ, IN_PAR_FEAT, IN_PAR_XYZ,
  IN_DC1_B, IN_DC1_W, IN_DC2_B, IN_DC2_W,
  IN_FD1_B, IN_FD1_W, IN_FD2_B, IN_FD2_W,
  IN_P1C1_B, IN_P1C1_W, IN_P1C2_B, IN_P1C2_W, IN_P1SC_B, IN_P1SC_W,
  IN_P2C1_B, IN_P2C1_W, IN_P2C2_B, IN_P2C2_W, IN_P2SC_B, IN_P2SC_W,
  IN_A1_B, IN_A1_W, IN_A2_B, IN_A2_W, IN_ABN_BT, IN_ABN_G,
  IN_END_B, IN_END_W, IN_K_B, IN_K_W,
  IN_POS1_B, IN_POS1_W, IN_POS2_B, IN_POS2_W, IN_POSBN_BT, IN_POSBN_G,
  IN_Q_B, IN_Q_W, IN_V_B, IN_V_W, IN_XYZ
};

static inline void gemm(hipStream_t s, bool acontig,
    const __bf16* A, long ars, long acs, long abs_,
    const __bf16* Bm, long brs, long bbs,
    float* C, long crs, long cbs,
    __bf16* Cbf, long cbrs, long cbbs,
    const float* bias, const float* g, const float* bt,
    const float* mean, const float* rstd,
    float* ssum, float* sssq,
    int M, int Kp, long Ncols, int nb, int flags) {
  dim3 grid((unsigned)(Ncols / 16), (unsigned)((M + 15) / 16), (unsigned)nb);
  if (acontig)
    k_wmma_gemm<true><<<grid, dim3(32), 0, s>>>(A, ars, acs, abs_, Bm, brs, bbs,
                                                C, crs, cbs, Cbf, cbrs, cbbs,
                                                bias, g, bt, mean, rstd, ssum, sssq,
                                                M, Kp, flags);
  else
    k_wmma_gemm<false><<<grid, dim3(32), 0, s>>>(A, ars, acs, abs_, Bm, brs, bbs,
                                                 C, crs, cbs, Cbf, cbrs, cbbs,
                                                 bias, g, bt, mean, rstd, ssum, sssq,
                                                 M, Kp, flags);
}

#define ELT(kern, total, ...) \
  kern<<<(unsigned)(((total) + 255) / 256), 256, 0, stream>>>(__VA_ARGS__)

extern "C" void kernel_launch(void* const* d_in, const int* in_sizes, int n_in,
                              void* d_out, int out_size, void* d_ws, size_t ws_size,
                              hipStream_t stream) {
  (void)in_sizes; (void)n_in; (void)out_size; (void)ws_size;

  const float* xyz      = (const float*)d_in[IN_XYZ];
  const float* par_xyz  = (const float*)d_in[IN_PAR_XYZ];
  const float* par_feat = (const float*)d_in[IN_PAR_FEAT];
  const float* gen_feat = (const float*)d_in[IN_GEN_FEAT];

  // output regions
  float* out_xyz = (float*)d_out;                               // (B,3,N*UP)
  float* out_f2  = out_xyz + (long)BB * 3 * NN * UPF;           // (B,512,N)
  float* out_qr  = out_f2 + (long)BB * 2 * FEATC * NN;          // (B,515,N)

  // ---- workspace bump allocator (256B aligned) ----
  size_t off = 0;
  auto alloc = [&](size_t bytes) -> char* {
    off = (off + 255) & ~(size_t)255;
    char* p = (char*)d_ws + off;
    off += bytes;
    return p;
  };
  auto fbuf = [&](long n) { return (float*)alloc(sizeof(float) * n); };
  auto hbuf = [&](long n) { return (__bf16*)alloc(sizeof(__bf16) * n); };
  auto ibuf = [&](long n) { return (int*)alloc(sizeof(int) * n); };

  // bf16 weights (zero-padded to Op x Cp)
  __bf16* wb_dc1  = hbuf(256L * 512);
  __bf16* wb_dc2  = hbuf(16L * 256);      // 6 -> 16 rows
  __bf16* wb_fd1  = hbuf(256L * 512);
  __bf16* wb_fd2  = hbuf(512L * 256);
  __bf16* wb_p1c1 = hbuf(512L * QDIMP);   // 515 -> 544 cols
  __bf16* wb_p1c2 = hbuf(512L * 512);
  __bf16* wb_p1sc = hbuf(512L * QDIMP);
  __bf16* wb_p2c1 = hbuf(256L * 512);
  __bf16* wb_p2c2 = hbuf(256L * 256);
  __bf16* wb_p2sc = hbuf(256L * 512);
  __bf16* wb_a1   = hbuf((long)ATTNH * DIMC);
  __bf16* wb_a2   = hbuf((long)DIMC * ATTNH);
  __bf16* wb_end  = hbuf((long)FEATC * DIMC);
  __bf16* wb_k    = hbuf((long)DIMC * FEATC);
  __bf16* wb_pos1 = hbuf((long)POSH * POSKP);  // 3 -> 32 cols
  __bf16* wb_pos2 = hbuf((long)DIMC * POSH);
  __bf16* wb_q    = hbuf((long)DIMC * FEATC);
  __bf16* wb_v    = hbuf((long)DIMC * FEATC);

  // stats block (zeroed each call): [pos_sum 64][pos_ssq 64][a_sum 512][a_ssq 512]
  float* stats = fbuf(64 + 64 + 512 + 512 + 64 + 64 + 512 + 512);
  float* pos_sum = stats,        * pos_ssq = stats + 64;
  float* a_sum   = stats + 128,  * a_ssq   = stats + 640;
  float* pos_mean = stats + 1152, * pos_rstd = stats + 1216;
  float* a_mean   = stats + 1280, * a_rstd   = stats + 1792;

  int*   idx1 = ibuf((long)BB * NN * KI);
  float* w1   = fbuf((long)BB * NN * KI);
  float* par_interp  = fbuf((long)BB * FEATC * NN);
  __bf16* bf_pinterp = hbuf((long)BB * FEATC * NN);
  __bf16* bf_pfeat   = hbuf((long)BB * FEATC * NN);
  float* qn2 = fbuf((long)BB * NN);
  float* rn2 = fbuf((long)BB * NN);
  float* Sb  = fbuf((long)NN * NN);                 // reused per batch
  int*   idx2 = ibuf((long)BB * NN * KI);
  float* w2   = fbuf((long)BB * NN * KI);
  float* gen_interp = fbuf((long)BB * FEATC * NN);
  __bf16* bf_qraw = hbuf((long)BB * QDIMP * NN);    // padded rows
  __bf16* h1_bf   = hbuf((long)BB * 512 * NN);
  float*  p1out   = fbuf((long)BB * 512 * NN);
  __bf16* bf_p1   = hbuf((long)BB * 512 * NN);
  __bf16* h2_bf   = hbuf((long)BB * FEATC * NN);
  float*  f1      = fbuf((long)BB * FEATC * NN);
  __bf16* bf_f1   = hbuf((long)BB * FEATC * NN);
  float* qm = fbuf((long)BB * DIMC * NN);
  float* km = fbuf((long)BB * DIMC * NN);
  float* vm = fbuf((long)BB * DIMC * NN);
  int*   idxA = ibuf((long)BB * NN * NKNN);
  __bf16* posrel_bf = hbuf((long)BB * POSKP * NN * NKNN);  // padded rows
  __bf16* ph_bf = hbuf((long)BB * POSH * NN * NKNN);
  float*  pe    = fbuf((long)BB * DIMC * NN * NKNN);
  __bf16* qkpe_bf = hbuf((long)BB * DIMC * NN * NKNN);
  __bf16* ah_bf  = hbuf((long)ATTNH * NN * NKNN);   // reused per batch
  float*  logits = fbuf((long)DIMC * NN * NKNN);    // reused per batch
  float*  agg    = fbuf((long)BB * DIMC * NN);
  __bf16* bf_agg = hbuf((long)BB * DIMC * NN);
  float*  Hbuf   = fbuf((long)BB * FEATC * NN);
  __bf16* bf_f2  = hbuf((long)BB * 2 * FEATC * NN);
  __bf16* hfd_bf = hbuf((long)BB * FEATC * NN);
  float*  fdbuf  = fbuf((long)BB * 512 * NN);
  __bf16* bf_fdr = hbuf((long)BB * 512 * NN);
  __bf16* hdc_bf = hbuf((long)BB * FEATC * NN);
  float*  dp     = fbuf((long)BB * 6 * NN);

  // ---- zero BN stats (deterministic every call) ----
  (void)hipMemsetAsync(stats, 0, sizeof(float) * 1152, stream);

  // ---- convert weights to padded bf16 ----
  struct WC { const void* s; __bf16* d; int O, C, Op, Cp; } wc[] = {
    {d_in[IN_DC1_W],  wb_dc1,  256, 512, 256, 512},
    {d_in[IN_DC2_W],  wb_dc2,  6, 256, 16, 256},
    {d_in[IN_FD1_W],  wb_fd1,  256, 512, 256, 512},
    {d_in[IN_FD2_W],  wb_fd2,  512, 256, 512, 256},
    {d_in[IN_P1C1_W], wb_p1c1, 512, QDIM, 512, QDIMP},
    {d_in[IN_P1C2_W], wb_p1c2, 512, 512, 512, 512},
    {d_in[IN_P1SC_W], wb_p1sc, 512, QDIM, 512, QDIMP},
    {d_in[IN_P2C1_W], wb_p2c1, 256, 512, 256, 512},
    {d_in[IN_P2C2_W], wb_p2c2, 256, 256, 256, 256},
    {d_in[IN_P2SC_W], wb_p2sc, 256, 512, 256, 512},
    {d_in[IN_A1_W],  wb_a1,  ATTNH, DIMC, ATTNH, DIMC},
    {d_in[IN_A2_W],  wb_a2,  DIMC, ATTNH, DIMC, ATTNH},
    {d_in[IN_END_W], wb_end, FEATC, DIMC, FEATC, DIMC},
    {d_in[IN_K_W],   wb_k,   DIMC, FEATC, DIMC, FEATC},
    {d_in[IN_POS1_W], wb_pos1, POSH, 3, POSH, POSKP},
    {d_in[IN_POS2_W], wb_pos2, DIMC, POSH, DIMC, POSH},
    {d_in[IN_Q_W],   wb_q,   DIMC, FEATC, DIMC, FEATC},
    {d_in[IN_V_W],   wb_v,   DIMC, FEATC, DIMC, FEATC},
  };
  for (auto& c : wc)
    ELT(k_cvt_pad, (long)c.Op * c.Cp, (const float*)c.s, c.d, c.O, c.C, c.Op, c.Cp);
  ELT(k_cvt, (long)BB * FEATC * NN, par_feat, bf_pfeat, (long)BB * FEATC * NN, 0);

  // ---- interp #1: coordinate kNN (xyz -> par_xyz) + IDW gather of par_feat ----
  ELT(k_knn3, (long)BB * NN, xyz, par_xyz, idx1, w1, KI, 1);
  ELT(k_interp, (long)BB * FEATC * NN, par_feat, idx1, w1, par_interp, FEATC);
  ELT(k_cvt, (long)BB * FEATC * NN, par_interp, bf_pinterp, (long)BB * FEATC * NN, 0);

  // ---- interp #2: feature-space kNN via WMMA Gram matrix, per batch ----
  ELT(k_rownorm, (long)BB * NN, par_interp, qn2, FEATC);
  ELT(k_rownorm, (long)BB * NN, par_feat, rn2, FEATC);
  for (int b = 0; b < BB; ++b) {
    const __bf16* Qb = bf_pinterp + (long)b * FEATC * NN;
    const __bf16* Rb = bf_pfeat + (long)b * FEATC * NN;
    // S[q,r] = sum_c Q[c,q]*R[c,r] : A is k-major (strided), k stride = NN
    gemm(stream, false, Qb, 0, NN, 0, Rb, NN, 0, Sb, NN, 0, nullptr, 0, 0,
         nullptr, nullptr, nullptr, nullptr, nullptr, nullptr, nullptr,
         NN, FEATC, NN, 1, 0);
    ELT(k_topk_S, (long)NN, Sb, qn2 + (long)b * NN, rn2 + (long)b * NN,
        idx2 + (long)b * NN * KI, w2 + (long)b * NN * KI);
  }
  ELT(k_interp, (long)BB * FEATC * NN, gen_feat, idx2, w2, gen_interp, FEATC);

  // ---- q_raw (output #3) ----
  ELT(k_concat_qraw, (long)BB * QDIMP * NN, xyz, par_interp, gen_interp, out_qr, bf_qraw);

  // ---- proj1 (mlp_res 515->512->512 + shortcut) ----
  gemm(stream, true, wb_p1c1, QDIMP, 0, 0, bf_qraw, NN, (long)QDIMP * NN,
       nullptr, 0, 0, h1_bf, NN, 512L * NN,
       (const float*)d_in[IN_P1C1_B], nullptr, nullptr, nullptr, nullptr,
       nullptr, nullptr, 512, QDIMP, NN, BB, 1);
  gemm(stream, true, wb_p1c2, 512, 0, 0, h1_bf, NN, 512L * NN,
       p1out, NN, 512L * NN, nullptr, 0, 0,
       (const float*)d_in[IN_P1C2_B], nullptr, nullptr, nullptr, nullptr,
       nullptr, nullptr, 512, 512, NN, BB, 0);
  gemm(stream, true, wb_p1sc, QDIMP, 0, 0, bf_qraw, NN, (long)QDIMP * NN,
       p1out, NN, 512L * NN, nullptr, 0, 0,
       (const float*)d_in[IN_P1SC_B], nullptr, nullptr, nullptr, nullptr,
       nullptr, nullptr, 512, QDIMP, NN, BB, 2);
  ELT(k_cvt, (long)BB * 512 * NN, p1out, bf_p1, (long)BB * 512 * NN, 0);

  // ---- proj2 (mlp_res 512->256->256 + shortcut) -> f1 ----
  gemm(stream, true, wb_p2c1, 512, 0, 0, bf_p1, NN, 512L * NN,
       nullptr, 0, 0, h2_bf, NN, (long)FEATC * NN,
       (const float*)d_in[IN_P2C1_B], nullptr, nullptr, nullptr, nullptr,
       nullptr, nullptr, FEATC, 512, NN, BB, 1);
  gemm(stream, true, wb_p2c2, FEATC, 0, 0, h2_bf, NN, (long)FEATC * NN,
       f1, NN, (long)FEATC * NN, nullptr, 0, 0,
       (const float*)d_in[IN_P2C2_B], nullptr, nullptr, nullptr, nullptr,
       nullptr, nullptr, FEATC, FEATC, NN, BB, 0);
  gemm(stream, true, wb_p2sc, 512, 0, 0, bf_p1, NN, 512L * NN,
       f1, NN, (long)FEATC * NN, nullptr, 0, 0,
       (const float*)d_in[IN_P2SC_B], nullptr, nullptr, nullptr, nullptr,
       nullptr, nullptr, FEATC, 512, NN, BB, 2);
  ELT(k_cvt, (long)BB * FEATC * NN, f1, bf_f1, (long)BB * FEATC * NN, 0);

  // ---- LCA: q/k/v projections ----
  gemm(stream, true, wb_q, FEATC, 0, 0, bf_f1, NN, (long)FEATC * NN,
       qm, NN, (long)DIMC * NN, nullptr, 0, 0,
       (const float*)d_in[IN_Q_B], nullptr, nullptr, nullptr, nullptr,
       nullptr, nullptr, DIMC, FEATC, NN, BB, 0);
  gemm(stream, true, wb_k, FEATC, 0, 0, bf_f1, NN, (long)FEATC * NN,
       km, NN, (long)DIMC * NN, nullptr, 0, 0,
       (const float*)d_in[IN_K_B], nullptr, nullptr, nullptr, nullptr,
       nullptr, nullptr, DIMC, FEATC, NN, BB, 0);
  gemm(stream, true, wb_v, FEATC, 0, 0, bf_f1, NN, (long)FEATC * NN,
       vm, NN, (long)DIMC * NN, nullptr, 0, 0,
       (const float*)d_in[IN_V_B], nullptr, nullptr, nullptr, nullptr,
       nullptr, nullptr, DIMC, FEATC, NN, BB, 0);

  // ---- attention kNN (self) + pos encoding branch (two-pass BN) ----
  ELT(k_knn3, (long)BB * NN, xyz, xyz, idxA, nullptr, NKNN, 0);
  ELT(k_posrel, (long)BB * POSKP * NN * NKNN, xyz, idxA, posrel_bf);
  const long NKcols = (long)NN * NKNN;  // 32768
  // pos1 pass1: stats only
  gemm(stream, true, wb_pos1, POSKP, 0, 0, posrel_bf, NKcols, (long)POSKP * NKcols,
       nullptr, 0, 0, nullptr, 0, 0,
       (const float*)d_in[IN_POS1_B], nullptr, nullptr, nullptr, nullptr,
       pos_sum, pos_ssq, POSH, POSKP, NKcols, BB, 0);
  k_finalize<<<1, 64, 0, stream>>>(pos_sum, pos_ssq, pos_mean, pos_rstd,
                                   POSH, (float)((long)BB * NN * NKNN));
  // pos1 pass2: BN + relu -> bf16
  gemm(stream, true, wb_pos1, POSKP, 0, 0, posrel_bf, NKcols, (long)POSKP * NKcols,
       nullptr, 0, 0, ph_bf, NKcols, (long)POSH * NKcols,
       (const float*)d_in[IN_POS1_B], (const float*)d_in[IN_POSBN_G],
       (const float*)d_in[IN_POSBN_BT], pos_mean, pos_rstd,
       nullptr, nullptr, POSH, POSKP, NKcols, BB, 1);
  // pos2 -> pe (f32)
  gemm(stream, true, wb_pos2, POSH, 0, 0, ph_bf, NKcols, (long)POSH * NKcols,
       pe, NKcols, (long)DIMC * NKcols, nullptr, 0, 0,
       (const float*)d_in[IN_POS2_B], nullptr, nullptr, nullptr, nullptr,
       nullptr, nullptr, DIMC, POSH, NKcols, BB, 0);

  // ---- attention MLP (two-pass BN), per-batch tail ----
  ELT(k_qkpe, (long)BB * DIMC * NN * NKNN, qm, km, idxA, pe, qkpe_bf);
  gemm(stream, true, wb_a1, DIMC, 0, 0, qkpe_bf, NKcols, (long)DIMC * NKcols,
       nullptr, 0, 0, nullptr, 0, 0,
       (const float*)d_in[IN_A1_B], nullptr, nullptr, nullptr, nullptr,
       a_sum, a_ssq, ATTNH, DIMC, NKcols, BB, 0);
  k_finalize<<<2, 256, 0, stream>>>(a_sum, a_ssq, a_mean, a_rstd,
                                    ATTNH, (float)((long)BB * NN * NKNN));
  for (int b = 0; b < BB; ++b) {
    const __bf16* qkb = qkpe_bf + (long)b * DIMC * NKcols;
    gemm(stream, true, wb_a1, DIMC, 0, 0, qkb, NKcols, 0,
         nullptr, 0, 0, ah_bf, NKcols, 0,
         (const float*)d_in[IN_A1_B], (const float*)d_in[IN_ABN_G],
         (const float*)d_in[IN_ABN_BT], a_mean, a_rstd,
         nullptr, nullptr, ATTNH, DIMC, NKcols, 1, 1);
    gemm(stream, true, wb_a2, ATTNH, 0, 0, ah_bf, NKcols, 0,
         logits, NKcols, 0, nullptr, 0, 0,
         (const float*)d_in[IN_A2_B], nullptr, nullptr, nullptr, nullptr,
         nullptr, nullptr, DIMC, ATTNH, NKcols, 1, 0);
    ELT(k_smax_agg, (long)DIMC * NN, logits, vm + (long)b * DIMC * NN,
        pe + (long)b * DIMC * NKcols, idxA + (long)b * NN * NKNN,
        agg + (long)b * DIMC * NN);
  }
  ELT(k_cvt, (long)BB * DIMC * NN, agg, bf_agg, (long)BB * DIMC * NN, 0);

  // ---- end conv + residual, f2 (output #2) ----
  gemm(stream, true, wb_end, DIMC, 0, 0, bf_agg, NN, (long)DIMC * NN,
       Hbuf, NN, (long)FEATC * NN, nullptr, 0, 0,
       (const float*)d_in[IN_END_B], nullptr, nullptr, nullptr, nullptr,
       nullptr, nullptr, FEATC, DIMC, NN, BB, 0);
  ELT(k_add, (long)BB * FEATC * NN, Hbuf, f1, (long)BB * FEATC * NN);
  ELT(k_concat_f2, (long)BB * 2 * FEATC * NN, f1, Hbuf, out_f2, bf_f2);

  // ---- feature decoder: fd = fd2(relu(fd1(f2))) ----
  gemm(stream, true, wb_fd1, 512, 0, 0, bf_f2, NN, 512L * NN,
       nullptr, 0, 0, hfd_bf, NN, (long)FEATC * NN,
       (const float*)d_in[IN_FD1_B], nullptr, nullptr, nullptr, nullptr,
       nullptr, nullptr, FEATC, 512, NN, BB, 1);
  gemm(stream, true, wb_fd2, FEATC, 0, 0, hfd_bf, NN, (long)FEATC * NN,
       fdbuf, NN, 512L * NN, nullptr, 0, 0,
       (const float*)d_in[IN_FD2_B], nullptr, nullptr, nullptr, nullptr,
       nullptr, nullptr, 512, FEATC, NN, BB, 0);
  ELT(k_cvt, (long)BB * 512 * NN, fdbuf, bf_fdr, (long)BB * 512 * NN, 1);  // relu(fd)

  // ---- delta decoder: dp = dc2(relu(dc1(relu(fd)))) ----
  gemm(stream, true, wb_dc1, 512, 0, 0, bf_fdr, NN, 512L * NN,
       nullptr, 0, 0, hdc_bf, NN, (long)FEATC * NN,
       (const float*)d_in[IN_DC1_B], nullptr, nullptr, nullptr, nullptr,
       nullptr, nullptr, FEATC, 512, NN, BB, 1);
  gemm(stream, true, wb_dc2, FEATC, 0, 0, hdc_bf, NN, (long)FEATC * NN,
       dp, NN, 6L * NN, nullptr, 0, 0,
       (const float*)d_in[IN_DC2_B], nullptr, nullptr, nullptr, nullptr,
       nullptr, nullptr, 6, FEATC, NN, BB, 0);

  // ---- point shuffle + upsample (output #1) ----
  ELT(k_shuffle, (long)BB * 3 * NN * UPF, xyz, dp, out_xyz);
}
